// HierarchicalGNN_51376398795544
// MI455X (gfx1250) — matrix operand order repaired and checked by reading, not compiled, not run on hardware
//
#include <hip/hip_runtime.h>
#include <hip/hip_bf16.h>
#include <math.h>

// ---------------- constants (match reference) ----------------
#define DHID   512
#define NHEAD  4
#define CHEAD  128
#define NGRAPH 64
#define NLABEL 32
#define HHID   256
#define BN_SCALE 0.9999950000374996f   // 1/sqrt(1 + 1e-5)

// ---------------- WMMA types ----------------
typedef __bf16 bf16_t;
typedef bf16_t v16bf __attribute__((ext_vector_type(16)));
typedef float  v8f   __attribute__((ext_vector_type(8)));
typedef unsigned short v8us __attribute__((ext_vector_type(8)));  // 16B LDS chunk

__device__ __forceinline__ unsigned f2bf_bits(float f) {
  unsigned u = __float_as_uint(f);
  u += 0x7FFFu + ((u >> 16) & 1u);      // round-to-nearest-even
  return u >> 16;
}
// monotone float <-> uint mapping for atomicMax on signed floats
__device__ __forceinline__ unsigned fkey(float f) {
  unsigned u = __float_as_uint(f);
  return (u & 0x80000000u) ? ~u : (u | 0x80000000u);
}
__device__ __forceinline__ float fkeyinv(unsigned u) {
  return __uint_as_float((u & 0x80000000u) ? (u ^ 0x80000000u) : ~u);
}
#define FKEY_NEGINF 0x007FFFFFu

// =====================================================================
// Tiled GEMM: C[M,N] = epilogue(A[M,K] @ B[K,N])   (bf16 compute, f32 acc)
// 64x64 macro tile, 4 waves; wave w owns rows [w*16,w*16+16), 4 n-subtiles.
// LDS tiles are stored in fragment-native order so every fragment is two
// 16B-aligned, bank-conflict-free ds_load_b128s.
//   A row K-order: [0..7, 16..23, 8..15, 24..31]  (groups 1 and 2 swapped)
//   B stored column-major: Bcol[n][k]
// act: 0=none 1=relu 2=tanh ; bnG!=null applies g*y*BN_SCALE+b after act.
// =====================================================================
__global__ __launch_bounds__(128) void gemm_wmma_bf16(
    const float* __restrict__ A, const float* __restrict__ B,
    float* __restrict__ C, int M, int N, int K,
    const float* __restrict__ bias, int act,
    const float* __restrict__ bnG, const float* __restrict__ bnB)
{
  __shared__ unsigned int   Asm[64][20];   // 64 rows x 40 u16 (32 + pad 8)
  __shared__ unsigned short Bcol[64][40];  // 64 cols x 40 u16 (32 + pad 8)
  const int tid  = threadIdx.x;
  const int lane = tid & 31, wave = tid >> 5;
  const int half = lane >> 4, lr = lane & 15;
  const int m0 = blockIdx.y * 64, n0 = blockIdx.x * 64;
  v8f acc[4] = {};

  for (int k0 = 0; k0 < K; k0 += 32) {
    // ---- stage A tile (64x32) as bf16, K-groups [0,2,1,3], packed u32 ----
    for (int i = tid; i < 64 * 16; i += 128) {      // 1024 pairs
      const int r  = i >> 4;                        // tile row 0..63
      const int pc = (i & 15) * 2;                  // source K pair 0,2,..,30
      const int gr = m0 + r;
      float2 v2 = make_float2(0.f, 0.f);
      if (gr < M) v2 = *(const float2*)(A + (size_t)gr * K + k0 + pc);
      const int g  = pc >> 3;
      const int dg = (g == 1) ? 2 : (g == 2) ? 1 : g;  // swap middle groups
      const int q  = (dg << 2) | ((pc & 7) >> 1);      // u32 slot in row
      Asm[r][q] = f2bf_bits(v2.x) | (f2bf_bits(v2.y) << 16);
    }
    // ---- stage B tile (32x64) column-major bf16 ----
    for (int i = tid; i < 32 * 64; i += 128) {
      const int r = i >> 6, c = i & 63;             // r = K row, c = col
      Bcol[c][r] = (unsigned short)f2bf_bits(B[(size_t)(k0 + r) * N + (n0 + c)]);
    }
    __syncthreads();

    // A fragment: two aligned 16B LDS loads
    union { v16bf v; v8us h[2]; } af;
    {
      const unsigned short* arow =
          (const unsigned short*)&Asm[wave * 16 + lr][0] + half * 16;
      af.h[0] = *(const v8us*)(arow);
      af.h[1] = *(const v8us*)(arow + 8);
    }
    #pragma unroll
    for (int nt = 0; nt < 4; ++nt) {
      union { v16bf v; v8us h[2]; } bfr;
      const unsigned short* bc = &Bcol[nt * 16 + lr][0] + half * 8;
      bfr.h[0] = *(const v8us*)(bc);
      bfr.h[1] = *(const v8us*)(bc + 16);
      acc[nt] = __builtin_amdgcn_wmma_f32_16x16x32_bf16(
          false, af.v, false, bfr.v, (short)0, acc[nt], false, false);
    }
    __syncthreads();
  }

  // epilogue; C/D layout: elem v -> row (v + 8*half), col lr within tile
  #pragma unroll
  for (int nt = 0; nt < 4; ++nt) {
    const int n = n0 + nt * 16 + lr;
    const float bv = bias ? bias[n] : 0.f;
    const float sg = bnG ? bnG[n] : 0.f;
    const float sb = bnB ? bnB[n] : 0.f;
    #pragma unroll
    for (int v = 0; v < 8; ++v) {
      const int m = m0 + wave * 16 + half * 8 + v;
      if (m < M) {
        float y = acc[nt][v] + bv;
        if (act == 1)      y = fmaxf(y, 0.f);
        else if (act == 2) y = tanhf(y);
        if (bnG) y = sg * y * BN_SCALE + sb;
        C[(size_t)m * N + n] = y;
      }
    }
  }
}

// =====================================================================
// Utility / elementwise kernels
// =====================================================================
__global__ void fill_u32_kernel(unsigned* p, unsigned v, int n) {
  for (int i = blockIdx.x * blockDim.x + threadIdx.x; i < n;
       i += gridDim.x * blockDim.x) p[i] = v;
}
__global__ void add_bias_rows(float* h, const float* __restrict__ bias, int n) {
  for (int i = blockIdx.x * blockDim.x + threadIdx.x; i < n;
       i += gridDim.x * blockDim.x) h[i] += bias[i & (DHID - 1)];
}
__global__ void elt_add(float* out, const float* __restrict__ a,
                        const float* __restrict__ b, int n) {
  for (int i = blockIdx.x * blockDim.x + threadIdx.x; i < n;
       i += gridDim.x * blockDim.x) out[i] = a[i] + b[i];
}

// =====================================================================
// GATv2 edge kernels (wave per edge; lane covers dims d = lane + 32*j)
// =====================================================================
__global__ __launch_bounds__(256) void gat_edge_score(
    const float* __restrict__ xl, const float* __restrict__ xr,
    const float* __restrict__ att, const int* __restrict__ src,
    const int* __restrict__ dst, float* __restrict__ score,
    unsigned* __restrict__ mU, int E)
{
  const int lane = threadIdx.x & 31, wave = threadIdx.x >> 5;
  const int e = blockIdx.x * 8 + wave;
  if (e >= E) return;
  const int sn = src[e], dn = dst[e];
  const float* pl = xl + (size_t)sn * DHID;
  const float* pr = xr + (size_t)dn * DHID;
  float hacc[4] = {0.f, 0.f, 0.f, 0.f};
  #pragma unroll
  for (int j = 0; j < 16; ++j) {
    const int d = lane + 32 * j;           // head = j>>2 (128 dims/head)
    float v = pl[d] + pr[d];
    v = (v > 0.f) ? v : 0.2f * v;          // leaky_relu 0.2
    hacc[j >> 2] += v * att[d];
  }
  #pragma unroll
  for (int h = 0; h < 4; ++h) {
    float v = hacc[h];
    for (int o = 16; o > 0; o >>= 1) v += __shfl_down(v, o, 32);
    if (lane == 0) {
      score[(size_t)e * 4 + h] = v;
      atomicMax(&mU[(size_t)dn * 4 + h], fkey(v));
    }
  }
}

__global__ void gat_edge_exp(float* __restrict__ score,
                             const unsigned* __restrict__ mU,
                             float* __restrict__ ssum,
                             const int* __restrict__ dst, int EH)
{
  for (int i = blockIdx.x * blockDim.x + threadIdx.x; i < EH;
       i += gridDim.x * blockDim.x) {
    const int e = i >> 2, h = i & 3;
    const int dn = dst[e];
    const float ex = expf(score[i] - fkeyinv(mU[(size_t)dn * 4 + h]));
    score[i] = ex;                          // reuse buffer as ex
    atomicAdd(&ssum[(size_t)dn * 4 + h], ex);
  }
}

__global__ __launch_bounds__(256) void gat_edge_aggregate(
    const float* __restrict__ ex, const float* __restrict__ ssum,
    const float* __restrict__ xl, const int* __restrict__ src,
    const int* __restrict__ dst, float* __restrict__ out, int E)
{
  const int lane = threadIdx.x & 31, wave = threadIdx.x >> 5;
  const int e = blockIdx.x * 8 + wave;
  if (e >= E) return;
  const int sn = src[e], dn = dst[e];
  const float* pl = xl + (size_t)sn * DHID;
  float* po = out + (size_t)dn * DHID;
  float alpha[4];
  #pragma unroll
  for (int h = 0; h < 4; ++h)
    alpha[h] = ex[(size_t)e * 4 + h] / (ssum[(size_t)dn * 4 + h] + 1e-16f);
  #pragma unroll
  for (int j = 0; j < 16; ++j) {
    const int d = lane + 32 * j;
    atomicAdd(&po[d], alpha[j >> 2] * pl[d]);
  }
}

// GIN neighbor sum: agg[dst] += hgat[src]
__global__ __launch_bounds__(256) void edge_agg_sum(
    const float* __restrict__ hgat, const int* __restrict__ src,
    const int* __restrict__ dst, float* __restrict__ agg, int E)
{
  const int lane = threadIdx.x & 31, wave = threadIdx.x >> 5;
  const int e = blockIdx.x * 8 + wave;
  if (e >= E) return;
  const float* ps = hgat + (size_t)src[e] * DHID;
  float* pd = agg + (size_t)dst[e] * DHID;
  #pragma unroll
  for (int j = 0; j < 16; ++j) {
    const int d = lane + 32 * j;
    atomicAdd(&pd[d], ps[d]);
  }
}

// =====================================================================
// Global LayerNorm (whole-tensor) reduction
// =====================================================================
__global__ void reduce_sums(const float* __restrict__ x, int n,
                            float* __restrict__ sums)
{
  __shared__ float sh0[256], sh1[256];
  float s0 = 0.f, s1 = 0.f;
  for (int i = blockIdx.x * blockDim.x + threadIdx.x; i < n;
       i += gridDim.x * blockDim.x) {
    const float v = x[i]; s0 += v; s1 += v * v;
  }
  sh0[threadIdx.x] = s0; sh1[threadIdx.x] = s1;
  __syncthreads();
  for (int st = 128; st > 0; st >>= 1) {
    if (threadIdx.x < st) {
      sh0[threadIdx.x] += sh0[threadIdx.x + st];
      sh1[threadIdx.x] += sh1[threadIdx.x + st];
    }
    __syncthreads();
  }
  if (threadIdx.x == 0) { atomicAdd(&sums[0], sh0[0]); atomicAdd(&sums[1], sh1[0]); }
}
__global__ void finalize_ln(float* sums, float invcount) {
  const float mu  = sums[0] * invcount;
  const float var = sums[1] * invcount - mu * mu;
  sums[2] = mu;
  sums[3] = rsqrtf(var + 1e-5f);
}
// h_out = leaky( (gin-mu)*inv*ln_g + ln_be + resout , 0.2 )
__global__ void layer_final(float* __restrict__ hout,
                            const float* __restrict__ gin,
                            const float* __restrict__ resout,
                            const float* __restrict__ lng,
                            const float* __restrict__ lnbe,
                            const float* __restrict__ stats, int n)
{
  const float mu = stats[2], inv = stats[3];
  for (int i = blockIdx.x * blockDim.x + threadIdx.x; i < n;
       i += gridDim.x * blockDim.x) {
    const int d = i & (DHID - 1);
    float y = (gin[i] - mu) * inv * lng[d] + lnbe[d] + resout[i];
    hout[i] = (y > 0.f) ? y : 0.2f * y;
  }
}

// =====================================================================
// Global-attention pooling
// =====================================================================
__global__ __launch_bounds__(256) void rowdot512(
    const float* __restrict__ t1, const float* __restrict__ w2,
    const float* __restrict__ b2, float* __restrict__ gate, int N)
{
  const int lane = threadIdx.x & 31, wave = threadIdx.x >> 5;
  const int n = blockIdx.x * 8 + wave;
  if (n >= N) return;
  const float* p = t1 + (size_t)n * DHID;
  float s = 0.f;
  #pragma unroll
  for (int j = 0; j < 16; ++j) { const int d = lane + 32 * j; s += p[d] * w2[d]; }
  for (int o = 16; o > 0; o >>= 1) s += __shfl_down(s, o, 32);
  if (lane == 0) gate[n] = s + b2[0];
}
__global__ void pool_max(const float* __restrict__ gate,
                         const int* __restrict__ batch,
                         unsigned* __restrict__ gmU, int N)
{
  for (int n = blockIdx.x * blockDim.x + threadIdx.x; n < N;
       n += gridDim.x * blockDim.x)
    atomicMax(&gmU[batch[n]], fkey(gate[n]));
}
__global__ void pool_exp(const float* __restrict__ gate,
                         const int* __restrict__ batch,
                         const unsigned* __restrict__ gmU,
                         float* __restrict__ ge, float* __restrict__ gs, int N)
{
  for (int n = blockIdx.x * blockDim.x + threadIdx.x; n < N;
       n += gridDim.x * blockDim.x) {
    const int b = batch[n];
    const float v = expf(gate[n] - fkeyinv(gmU[b]));
    ge[n] = v;
    atomicAdd(&gs[b], v);
  }
}
__global__ void pool_emb(const float* __restrict__ ge,
                         const float* __restrict__ gs,
                         const int* __restrict__ batch,
                         const float* __restrict__ h,
                         float* __restrict__ emb, int ND)
{
  for (int i = blockIdx.x * blockDim.x + threadIdx.x; i < ND;
       i += gridDim.x * blockDim.x) {
    const int n = i >> 9, d = i & (DHID - 1);
    const int b = batch[n];
    const float a = ge[n] / (gs[b] + 1e-16f);
    atomicAdd(&emb[(size_t)b * DHID + d], a * h[i]);
  }
}

// =====================================================================
// Label heads: block per (graph g, label l); thread j owns hidden unit j
// =====================================================================
__global__ __launch_bounds__(256) void heads_kernel(
    const float* __restrict__ emb, const float* __restrict__ W1,
    const float* __restrict__ b1, const float* __restrict__ g,
    const float* __restrict__ be, const float* __restrict__ W2,
    const float* __restrict__ b2, float* __restrict__ out)
{
  const int gi = blockIdx.x, l = blockIdx.y, j = threadIdx.x;
  const float* e  = emb + (size_t)gi * DHID;
  const float* w1 = W1 + (size_t)l * DHID * HHID;
  float acc = b1[l * HHID + j];
  for (int k = 0; k < DHID; ++k) acc += e[k] * w1[(size_t)k * HHID + j];
  float y = acc / (1.f + expf(-acc));                 // silu
  y = g[l * HHID + j] * y * BN_SCALE + be[l * HHID + j];
  __shared__ float red[256];
  red[j] = y * W2[l * HHID + j];
  __syncthreads();
  for (int st = 128; st > 0; st >>= 1) {
    if (j < st) red[j] += red[j + st];
    __syncthreads();
  }
  if (j == 0) out[gi * NLABEL + l] = red[0] + b2[l];
}

// =====================================================================
// Host launcher
// =====================================================================
extern "C" void kernel_launch(void* const* d_in, const int* in_sizes, int n_in,
                              void* d_out, int out_size, void* d_ws, size_t ws_size,
                              hipStream_t stream)
{
  const float* x       = (const float*)d_in[0];
  const int*   eidx    = (const int*)d_in[1];
  const int*   batch   = (const int*)d_in[2];
  const float* proj_W  = (const float*)d_in[3];
  const float* proj_b  = (const float*)d_in[4];
  const float* proj_g  = (const float*)d_in[5];
  const float* proj_be = (const float*)d_in[6];
  const float* gat_Wl  = (const float*)d_in[7];
  const float* gat_Wr  = (const float*)d_in[8];
  const float* gat_att = (const float*)d_in[9];
  const float* gat_b   = (const float*)d_in[10];
  const float* gin_W   = (const float*)d_in[11];
  const float* gin_b   = (const float*)d_in[12];
  const float* gin_g   = (const float*)d_in[13];
  const float* gin_be  = (const float*)d_in[14];
  const float* ln_g    = (const float*)d_in[15];
  const float* ln_be   = (const float*)d_in[16];
  const float* res_W   = (const float*)d_in[17];
  const float* res_b   = (const float*)d_in[18];
  const float* gate_W1 = (const float*)d_in[19];
  const float* gate_b1 = (const float*)d_in[20];
  const float* gate_W2 = (const float*)d_in[21];
  const float* gate_b2 = (const float*)d_in[22];
  const float* head_W1 = (const float*)d_in[23];
  const float* head_b1 = (const float*)d_in[24];
  const float* head_g  = (const float*)d_in[25];
  const float* head_be = (const float*)d_in[26];
  const float* head_W2 = (const float*)d_in[27];
  const float* head_b2 = (const float*)d_in[28];

  const int N   = in_sizes[2];
  const int E   = in_sizes[1] / 2;
  const int DIN = in_sizes[0] / N;
  const int D   = DHID;
  const int T   = in_sizes[10] / D;     // gat_b is (T, D)
  const int ND  = N * D;
  const int* src = eidx;
  const int* dst = eidx + E;

  // ---- workspace carve ----
  char* w = (char*)d_ws;
  size_t off = 0;
  auto carve = [&](size_t bytes) -> void* {
    void* p = w + off;
    off = (off + bytes + 255) & ~(size_t)255;
    return p;
  };
  float*    B0    = (float*)carve((size_t)ND * 4);   // h / res_in
  float*    B1    = (float*)carve((size_t)ND * 4);   // xl / gin-in / t1
  float*    B2    = (float*)carve((size_t)ND * 4);   // xr / agg / resout
  float*    B3    = (float*)carve((size_t)ND * 4);   // gat out / gin out
  float*    score = (float*)carve((size_t)E * NHEAD * 4);
  unsigned* mU    = (unsigned*)carve((size_t)N * NHEAD * 4);
  float*    ssum  = (float*)carve((size_t)N * NHEAD * 4);
  float*    gate  = (float*)carve((size_t)N * 4);
  float*    ge    = (float*)carve((size_t)N * 4);
  unsigned* gmU   = (unsigned*)carve(NGRAPH * 4);
  float*    gs    = (float*)carve(NGRAPH * 4);
  float*    emb   = (float*)carve((size_t)NGRAPH * D * 4);
  float*    stats = (float*)carve(4 * 4);
  (void)ws_size; (void)n_in; (void)out_size;

  const int tpb = 256;
  int gsb = (ND + tpb - 1) / tpb; if (gsb > 8192) gsb = 8192;
  const int ebk = (E + 7) / 8;                 // wave-per-edge blocks
  const int nbk = (N + tpb - 1) / tpb;

  auto gemm = [&](const float* A, const float* Bm, float* Cm, int M, int Nn,
                  int K, const float* bias, int act, const float* bg,
                  const float* bb) {
    dim3 grid(Nn / 64, (M + 63) / 64);
    gemm_wmma_bf16<<<grid, dim3(128), 0, stream>>>(A, Bm, Cm, M, Nn, K, bias,
                                                   act, bg, bb);
  };

  // ---- feature_proj: bn(relu(x @ W + b)) -> B0 ----
  gemm(x, proj_W, B0, N, D, DIN, proj_b, 1, proj_g, proj_be);

  // ---- layers ----
  for (int t = 0; t < T; ++t) {
    const float* Wl   = gat_Wl + (size_t)t * D * D;
    const float* Wr   = gat_Wr + (size_t)t * D * D;
    const float* attt = gat_att + (size_t)t * NHEAD * CHEAD;
    const float* gbt  = gat_b + (size_t)t * D;
    const float* gWt  = gin_W + (size_t)t * D * D;
    const float* gbbt = gin_b + (size_t)t * D;
    const float* ggt  = gin_g + (size_t)t * D;
    const float* gbet = gin_be + (size_t)t * D;
    const float* lgt  = ln_g + (size_t)t * D;
    const float* lbt  = ln_be + (size_t)t * D;
    const float* rWt  = res_W + (size_t)t * D * D;
    const float* rbt  = res_b + (size_t)t * D;

    // GATv2 projections
    gemm(B0, Wl, B1, N, D, D, nullptr, 0, nullptr, nullptr);   // xl
    gemm(B0, Wr, B2, N, D, D, nullptr, 0, nullptr, nullptr);   // xr

    // segment softmax state
    fill_u32_kernel<<<(N * NHEAD + tpb - 1) / tpb, tpb, 0, stream>>>(
        mU, FKEY_NEGINF, N * NHEAD);
    hipMemsetAsync(ssum, 0, (size_t)N * NHEAD * 4, stream);
    hipMemsetAsync(B3, 0, (size_t)ND * 4, stream);

    gat_edge_score<<<ebk, 256, 0, stream>>>(B1, B2, attt, src, dst, score, mU, E);
    gat_edge_exp<<<(E * NHEAD + tpb - 1) / tpb, tpb, 0, stream>>>(
        score, mU, ssum, dst, E * NHEAD);
    gat_edge_aggregate<<<ebk, 256, 0, stream>>>(score, ssum, B1, src, dst, B3, E);
    add_bias_rows<<<gsb, tpb, 0, stream>>>(B3, gbt, ND);       // h_gat in B3

    // GIN: agg (B2) then mlp((h+agg)) with fused relu+bn -> B3
    hipMemsetAsync(B2, 0, (size_t)ND * 4, stream);
    edge_agg_sum<<<ebk, 256, 0, stream>>>(B3, src, dst, B2, E);
    elt_add<<<gsb, tpb, 0, stream>>>(B1, B3, B2, ND);          // B1 = h_gat+agg
    gemm(B1, gWt, B3, N, D, D, gbbt, 1, ggt, gbet);            // gin out -> B3

    // global LayerNorm stats
    hipMemsetAsync(stats, 0, 16, stream);
    reduce_sums<<<4096, tpb, 0, stream>>>(B3, ND, stats);
    finalize_ln<<<1, 1, 0, stream>>>(stats, 1.0f / (float)ND);

    // residual path and fused finish -> new h in B0
    gemm(B0, rWt, B2, N, D, D, rbt, 0, nullptr, nullptr);      // resout -> B2
    layer_final<<<gsb, tpb, 0, stream>>>(B0, B3, B2, lgt, lbt, stats, ND);
  }

  // ---- GlobalAttention pooling ----
  gemm(B0, gate_W1, B1, N, D, D, gate_b1, 2, nullptr, nullptr);  // tanh
  rowdot512<<<(N + 7) / 8, 256, 0, stream>>>(B1, gate_W2, gate_b2, gate, N);
  fill_u32_kernel<<<1, 64, 0, stream>>>(gmU, FKEY_NEGINF, NGRAPH);
  hipMemsetAsync(gs, 0, NGRAPH * 4, stream);
  hipMemsetAsync(emb, 0, (size_t)NGRAPH * D * 4, stream);
  pool_max<<<nbk, tpb, 0, stream>>>(gate, batch, gmU, N);
  pool_exp<<<nbk, tpb, 0, stream>>>(gate, batch, gmU, ge, gs, N);
  pool_emb<<<gsb, tpb, 0, stream>>>(ge, gs, batch, B0, emb, ND);

  // ---- label heads ----
  heads_kernel<<<dim3(NGRAPH, NLABEL), 256, 0, stream>>>(
      emb, head_W1, head_b1, head_g, head_be, head_W2, head_b2, (float*)d_out);
}
